// MaskLM_45561013076167
// MI455X (gfx1250) — matrix operand order repaired and checked by reading, not compiled, not run on hardware
//
#include <hip/hip_runtime.h>

// Problem sizes (fixed by the reference)
#define B_  16
#define S_  512
#define H_  768
#define V_  30522
#define P_  77
#define M_  (B_ * P_)              // 1232 masked rows, = 77 M-tiles of 16
#define NT_ 4                      // N-tiles per wave -> 16x64 output per wave
#define COLS_PER_BLOCK_ (8 * 16 * NT_)   // 8 waves * 64 cols = 512

typedef __attribute__((ext_vector_type(16))) __bf16       v16bf;
typedef __attribute__((ext_vector_type(8)))  float        v8f;
typedef __attribute__((ext_vector_type(4)))  unsigned int v4u;

union FragBF {
    v16bf v;
    v4u   u[2];
};

__device__ __forceinline__ unsigned short f2bf(float f) {
    unsigned int x = __float_as_uint(f);
    x += 0x7FFFu + ((x >> 16) & 1u);      // round-to-nearest-even
    return (unsigned short)(x >> 16);
}

// ---------------------------------------------------------------------------
// Kernel 1: W2 fp32 -> bf16 (one-shot; bf16 W2 (47MB) then lives in 192MB L2)
// ---------------------------------------------------------------------------
__global__ void __launch_bounds__(256) convert_w2_bf16(
    const float* __restrict__ W2, unsigned short* __restrict__ W2b, int n4) {
    int i = blockIdx.x * 256 + threadIdx.x;
    if (i >= n4) return;
    const float4 f = reinterpret_cast<const float4*>(W2)[i];
    uint2 o;
    o.x = (unsigned int)f2bf(f.x) | ((unsigned int)f2bf(f.y) << 16);
    o.y = (unsigned int)f2bf(f.z) | ((unsigned int)f2bf(f.w) << 16);
    reinterpret_cast<uint2*>(W2b)[i] = o;
}

// ---------------------------------------------------------------------------
// Kernel 2: gather + (x @ W1^T + b1) + exact GELU + LayerNorm, emit bf16 rows.
// One block per masked row; fp32 throughout (only 2.5% of total FLOPs).
// ---------------------------------------------------------------------------
__global__ void __launch_bounds__(256) gather_mlp_ln(
    const float* __restrict__ X, const long long* __restrict__ pos,
    const float* __restrict__ W1, const float* __restrict__ b1,
    const float* __restrict__ gamma, const float* __restrict__ beta,
    unsigned short* __restrict__ Hn) {
    __shared__ __attribute__((aligned(16))) float xrow[H_];
    __shared__ float red[256], red2[256];

    const int row = blockIdx.x;             // 0..1231
    const int b   = row / P_;
    const int p   = row % P_;
    const int t   = threadIdx.x;

    const long long sp = pos[(size_t)b * P_ + p];
    const float* xsrc  = X + ((size_t)b * S_ + (size_t)sp) * H_;
    for (int k = t; k < H_; k += 256) xrow[k] = xsrc[k];
    __syncthreads();

    float hv[3];
    float s = 0.f, s2 = 0.f;
#pragma unroll
    for (int j = 0; j < 3; ++j) {
        const int o = t + 256 * j;
        const float4* w4 = reinterpret_cast<const float4*>(W1 + (size_t)o * H_);
        const float4* x4 = reinterpret_cast<const float4*>(xrow);
        float acc = b1[o];
#pragma unroll 8
        for (int k = 0; k < H_ / 4; ++k) {
            const float4 a = x4[k], w = w4[k];
            acc += a.x * w.x + a.y * w.y + a.z * w.z + a.w * w.w;
        }
        // exact GELU: 0.5*x*(1+erf(x/sqrt(2)))
        const float g = 0.5f * acc * (1.0f + erff(acc * 0.70710678118654752f));
        hv[j] = g;
        s += g; s2 += g * g;
    }
    red[t] = s; red2[t] = s2;
    __syncthreads();
    for (int off = 128; off > 0; off >>= 1) {
        if (t < off) { red[t] += red[t + off]; red2[t] += red2[t + off]; }
        __syncthreads();
    }
    const float mu   = red[0]  * (1.0f / H_);
    const float var  = red2[0] * (1.0f / H_) - mu * mu;
    const float rstd = rsqrtf(var + 1e-5f);

    unsigned short* dst = Hn + (size_t)row * H_;
#pragma unroll
    for (int j = 0; j < 3; ++j) {
        const int o = t + 256 * j;
        const float y = (hv[j] - mu) * rstd * gamma[o] + beta[o];
        dst[o] = f2bf(y);
    }
}

// Scheduler steering helpers (no-ops if the builtin is unavailable)
#if defined(__has_builtin)
#  if __has_builtin(__builtin_amdgcn_sched_group_barrier)
#    define SCHED_GROUP(mask, size) __builtin_amdgcn_sched_group_barrier(mask, size, 0)
#  endif
#endif
#ifndef SCHED_GROUP
#  define SCHED_GROUP(mask, size)
#endif
#define SG_WMMA_     0x008   // MFMA / WMMA
#define SG_VMEMREAD_ 0x020   // VMEM read

// ---------------------------------------------------------------------------
// Kernel 3: vocab projection, bf16 WMMA with f32 accumulate.
// logits[m, n] = sum_k Hn[m,k] * W2[n,k] + b2[n]
// Wave32: each wave owns a 16x64 output block (4 N-tiles), software-pipelined
// one K-step ahead (double-buffered A and B). amdgpu_waves_per_eu(1,4) lowers
// the occupancy target so the ~150-VGPR double buffer fits, and
// sched_group_barrier interleaves next-step loads between the 4 WMMAs.
// ---------------------------------------------------------------------------
__global__ void __launch_bounds__(256)
__attribute__((amdgpu_waves_per_eu(1, 4)))
vocab_gemm_wmma(
    const unsigned short* __restrict__ Hn, const unsigned short* __restrict__ W2b,
    const float* __restrict__ b2, float* __restrict__ out) {
    const int tid  = threadIdx.x;
    const int wave = tid >> 5;
    const int lane = tid & 31;
    const int lo   = lane & 15;
    const int hi   = lane >> 4;

    const int n_base = (blockIdx.x * 8 + wave) * (16 * NT_);
    if (n_base >= V_) return;                         // wave-uniform skip
    const int m_base = blockIdx.y * 16;               // M_ = 77*16 exact

    // A frag: lane lo -> row m_base+lo; half hi -> K subsets {hi*8..},{16+hi*8..}
    const unsigned short* arow = Hn + (size_t)(m_base + lo) * H_ + hi * 8;

    // Per N-tile column index + clamped B row pointers (B[k][n] = W2[n][k])
    int n_col[NT_];
    const unsigned short* brow[NT_];
#pragma unroll
    for (int j = 0; j < NT_; ++j) {
        const int n  = n_base + j * 16 + lo;
        n_col[j]     = n;
        const int nc = (n < V_) ? n : (V_ - 1);       // clamp ragged loads
        brow[j]      = W2b + (size_t)nc * H_ + hi * 8;
    }

    v8f acc[NT_];
#pragma unroll
    for (int j = 0; j < NT_; ++j)
        acc[j] = (v8f){0.f, 0.f, 0.f, 0.f, 0.f, 0.f, 0.f, 0.f};

    // -------- software pipeline: prologue loads for k = 0 --------
    FragBF aCur, aNxt;
    FragBF bCur[NT_], bNxt[NT_];
    aCur.u[0] = *reinterpret_cast<const v4u*>(arow);          // K = hi*8 .. +7
    aCur.u[1] = *reinterpret_cast<const v4u*>(arow + 16);     // K = 16+hi*8 .. +7
#pragma unroll
    for (int j = 0; j < NT_; ++j) {
        bCur[j].u[0] = *reinterpret_cast<const v4u*>(brow[j]);      // K = hi*8 ..
        bCur[j].u[1] = *reinterpret_cast<const v4u*>(brow[j] + 16); // K = 16+hi*8 ..
    }

#pragma unroll
    for (int k = 0; k < H_; k += 32) {
        const int kn = k + 32;
        if (kn < H_) {   // issue next step's loads before this step's WMMAs
            aNxt.u[0] = *reinterpret_cast<const v4u*>(arow + kn);
            aNxt.u[1] = *reinterpret_cast<const v4u*>(arow + kn + 16);
#pragma unroll
            for (int j = 0; j < NT_; ++j) {
                bNxt[j].u[0] = *reinterpret_cast<const v4u*>(brow[j] + kn);
                bNxt[j].u[1] = *reinterpret_cast<const v4u*>(brow[j] + kn + 16);
            }
        }
#pragma unroll
        for (int j = 0; j < NT_; ++j) {
            acc[j] = __builtin_amdgcn_wmma_f32_16x16x32_bf16(
                /*neg_a=*/false, aCur.v, /*neg_b=*/false, bCur[j].v,
                /*c_mod=*/(short)0, acc[j], /*reuse_a=*/false, /*reuse_b=*/false);
        }
        aCur = aNxt;
#pragma unroll
        for (int j = 0; j < NT_; ++j) bCur[j] = bNxt[j];

        // Interleave: next-step loads threaded between this step's WMMAs.
        if (kn < H_) {
            SCHED_GROUP(SG_WMMA_, 1);  SCHED_GROUP(SG_VMEMREAD_, 3);
            SCHED_GROUP(SG_WMMA_, 1);  SCHED_GROUP(SG_VMEMREAD_, 3);
            SCHED_GROUP(SG_WMMA_, 1);  SCHED_GROUP(SG_VMEMREAD_, 2);
            SCHED_GROUP(SG_WMMA_, 1);  SCHED_GROUP(SG_VMEMREAD_, 2);
        } else {
            SCHED_GROUP(SG_WMMA_, 4);
        }
    }

#pragma unroll
    for (int j = 0; j < NT_; ++j) {
        const int n = n_col[j];
        const float bias = b2[(n < V_) ? n : (V_ - 1)];  // same n for all 8 rows
#pragma unroll
        for (int r = 0; r < 8; ++r) acc[j][r] += bias;
        if (n < V_) {                                 // lane-masked stores only
#pragma unroll
            for (int r = 0; r < 8; ++r) {
                const int m = m_base + r + hi * 8;    // C layout: vgpr r, half hi
                out[(size_t)m * V_ + n] = acc[j][r];
            }
        }
    }
}

// ---------------------------------------------------------------------------
extern "C" void kernel_launch(void* const* d_in, const int* in_sizes, int n_in,
                              void* d_out, int out_size, void* d_ws, size_t ws_size,
                              hipStream_t stream) {
    const float*      X     = (const float*)d_in[0];
    const long long*  pp    = (const long long*)d_in[1];   // int64 positions
    const float*      W1    = (const float*)d_in[2];
    const float*      b1    = (const float*)d_in[3];
    const float*      gamma = (const float*)d_in[4];
    const float*      beta  = (const float*)d_in[5];
    const float*      W2    = (const float*)d_in[6];
    const float*      b2    = (const float*)d_in[7];
    float*            out   = (float*)d_out;

    // workspace: [ W2 bf16 : V*H ][ Hn bf16 : M*H ]  (~46.5 MB)
    unsigned short* W2b = (unsigned short*)d_ws;
    unsigned short* Hn  = W2b + (size_t)V_ * H_;

    const int n4 = (V_ * H_) / 4;
    convert_w2_bf16<<<(n4 + 255) / 256, 256, 0, stream>>>(W2, W2b, n4);

    gather_mlp_ln<<<M_, 256, 0, stream>>>(X, pp, W1, b1, gamma, beta, Hn);

    // 8 waves/block, each wave = 16x64 output; block covers 512 columns.
    dim3 grid((V_ + COLS_PER_BLOCK_ - 1) / COLS_PER_BLOCK_, M_ / 16);  // 60 x 77
    vocab_gemm_wmma<<<grid, 256, 0, stream>>>(Hn, W2b, b2, out);
}
